// InterAttention_10608569221506
// MI455X (gfx1250) — matrix operand order repaired and checked by reading, not compiled, not run on hardware
//
#include <hip/hip_runtime.h>
#include <hip/hip_bf16.h>

// ---------------------------------------------------------------------------
// InterAttention for MI455X (gfx1250, wave32, WMMA).
// bf16 A/B operands + f32 WMMA accumulation (16x16x32 bf16 = 8x MACs/instr of
// the f32 16x16x4 form); softmax stats and att matrix in f32 (att = 128 MB,
// fits the 192 MB L2). B-operand tiles staged in LDS via the gfx1250 async
// global->LDS path (ASYNCcnt) with double buffering when available.
// ---------------------------------------------------------------------------

typedef __bf16 bf16;
typedef __attribute__((ext_vector_type(16))) __bf16 v16bf;
typedef __attribute__((ext_vector_type(8)))  __bf16 v8bf;
typedef __attribute__((ext_vector_type(4)))  __bf16 v4bf;
typedef __attribute__((ext_vector_type(8)))  float  v8f;
typedef __attribute__((ext_vector_type(4)))  float  v4f;
typedef __attribute__((ext_vector_type(4)))  int    v4i;

#define B_  32
#define LP_ 2048
#define LQ_ 512
#define H_  1024

#define WMMA_BF16(a, b, c) \
    __builtin_amdgcn_wmma_f32_16x16x32_bf16(false, (a), false, (b), (short)0, (c), false, false)

// ---------------------------------------------------------------------------
// Async global->LDS (16B per lane). The gfx1250 builtin takes (global int4*,
// lds int4*, imm offset, imm cpol). Falls back to a sync LDS store if absent.
// ---------------------------------------------------------------------------
#if defined(__gfx1250__) && __has_builtin(__builtin_amdgcn_global_load_async_to_lds_b128)
#define HAVE_ASYNC_LDS 1
#else
#define HAVE_ASYNC_LDS 0
#endif

__device__ __forceinline__ void async_b128(const void* g, void* l) {
#if HAVE_ASYNC_LDS
    __builtin_amdgcn_global_load_async_to_lds_b128(
        (__attribute__((address_space(1))) v4i*)(g),
        (__attribute__((address_space(3))) v4i*)(l), 0, 0);
#else
    *(v4i*)l = *(const v4i*)g;
#endif
}

#if HAVE_ASYNC_LDS
#if __has_builtin(__builtin_amdgcn_s_wait_asynccnt)
#define WAIT_ASYNC(n) __builtin_amdgcn_s_wait_asynccnt(n)
#else
#define WAIT_ASYNC(n) asm volatile("s_wait_asynccnt %0" ::"i"(n) : "memory")
#endif
#else
#define WAIT_ASYNC(n) do {} while (0)
#endif

// ---------------------------------------------------------------------------
// f32 -> bf16 cast, 4 elements per thread
// ---------------------------------------------------------------------------
__global__ void __launch_bounds__(256) cast_f32_bf16(const float* __restrict__ src,
                                                     bf16* __restrict__ dst) {
    size_t i = ((size_t)blockIdx.x * blockDim.x + threadIdx.x) * 4;
    v4f x = *(const v4f*)(src + i);
    v4bf y;
#pragma unroll
    for (int j = 0; j < 4; ++j) y[j] = (bf16)x[j];
    *(v4bf*)(dst + i) = y;
}

// ---------------------------------------------------------------------------
// Fragment loaders (CDNA5 ISA 7.12.2, wave32)
// A 16x32 bf16: lane -> M = lane%16; element e -> K = 16*half + e (order used here, not the ISA table order)
// B 32x16 bf16: lane -> N = lane%16; element e -> K = 16*half + e (same order as A, not the ISA table order)
// C 16x16 f32 : lane -> N = lane%16; vgpr r  -> M = 8*half + r
// ---------------------------------------------------------------------------
__device__ __forceinline__ v16bf load_a_frag(const bf16* __restrict__ base, int ld,
                                             int row, int kk, int half) {
    const bf16* p0 = base + (size_t)row * ld + kk + 16 * half;
    v8bf x0 = *(const v8bf*)p0;
    v8bf x1 = *(const v8bf*)(p0 + 8);
    v16bf a;
#pragma unroll
    for (int i = 0; i < 8; ++i) { a[i] = x0[i]; a[i + 8] = x1[i]; }
    return a;
}

// 16 consecutive K values starting at p (K-major LDS row) -> B fragment.
__device__ __forceinline__ v16bf lds_b_frag_k(const bf16* p) {
    v8bf x0 = *(const v8bf*)p;
    v8bf x1 = *(const v8bf*)(p + 8);
    v16bf b;
#pragma unroll
    for (int i = 0; i < 8; ++i) { b[i] = x0[i]; b[i + 8] = x1[i]; }
    return b;
}

// Transposed read from [K][N]-layout LDS tile.
__device__ __forceinline__ v16bf lds_b_frag_t(const bf16 (&sB)[32][72], int nrel, int half) {
    v16bf bfrag;
#pragma unroll
    for (int e = 0; e < 16; ++e) bfrag[e] = sB[16 * half + e][nrel];
    return bfrag;
}

// A fragment from row-softmaxed att (K-contiguous f32 loads + exp + cvt)
__device__ __forceinline__ v16bf softmax_a_frag(const float* __restrict__ attRow, int kk,
                                                int half, float mx, float rc) {
    v8f x0 = *(const v8f*)(attRow + kk + 16 * half);
    v8f x1 = *(const v8f*)(attRow + kk + 16 * half + 8);
    v16bf a;
#pragma unroll
    for (int i = 0; i < 8; ++i) {
        a[i]     = (bf16)(__expf(x0[i] - mx) * rc);
        a[i + 8] = (bf16)(__expf(x1[i] - mx) * rc);
    }
    return a;
}

// ---------------------------------------------------------------------------
// Kernel: q_proj = q_encode @ G^T + bias   (M = B*LQ, N = H, K = H), bf16 out
// B tile (64 cols x 32 k, K-major, 80B row stride) async-staged in LDS,
// double buffered: fetch of step i+1 overlaps WMMAs of step i.
// ---------------------------------------------------------------------------
__global__ void __launch_bounds__(256) gemm_qproj(const bf16* __restrict__ Q,
                                                  const bf16* __restrict__ W,
                                                  const float* __restrict__ bias,
                                                  bf16* __restrict__ out) {
    __shared__ bf16 sB[2][64][40];
    const int lane = threadIdx.x & 31;
    const int wave = threadIdx.x >> 5;
    const int lo = lane & 15, half = lane >> 4;
    const int m0 = blockIdx.x * 128 + (wave & 3) * 32;
    const int nBlock = blockIdx.y * 64;
    const int nrel0 = (wave >> 2) * 32 + lo;

    const int sn = threadIdx.x >> 2;            // 0..63: B row (N)
    const int sk = (threadIdx.x & 3) * 8;       // 0..24: 8-elem (16B) K chunk

    auto stageB = [&](int buf, int kk) {
        async_b128(W + (size_t)(nBlock + sn) * H_ + kk + sk, &sB[buf][sn][sk]);
    };

    stageB(0, 0);
    v8f c[2][2] = {};
    for (int kk = 0, it = 0; kk < H_; kk += 32, ++it) {
        if (kk + 32 < H_) { stageB((it + 1) & 1, kk + 32); WAIT_ASYNC(1); }
        else              { WAIT_ASYNC(0); }
        __syncthreads();
        __builtin_prefetch(Q + (size_t)(m0 + lo) * H_ + kk + 64, 0, 3);
        const bf16 (&Bt)[64][40] = sB[it & 1];
        v16bf a0 = load_a_frag(Q, H_, m0 + lo, kk, half);
        v16bf a1 = load_a_frag(Q, H_, m0 + 16 + lo, kk, half);
        v16bf b0 = lds_b_frag_k(&Bt[nrel0][16 * half]);
        v16bf b1 = lds_b_frag_k(&Bt[nrel0 + 16][16 * half]);
        c[0][0] = WMMA_BF16(a0, b0, c[0][0]);
        c[0][1] = WMMA_BF16(a0, b1, c[0][1]);
        c[1][0] = WMMA_BF16(a1, b0, c[1][0]);
        c[1][1] = WMMA_BF16(a1, b1, c[1][1]);
        __syncthreads();
    }
    const int n0 = nBlock + (wave >> 2) * 32;
#pragma unroll
    for (int mi = 0; mi < 2; ++mi)
#pragma unroll
        for (int ni = 0; ni < 2; ++ni) {
            int n = n0 + ni * 16 + lo;
            float bs = bias[n];
#pragma unroll
            for (int r = 0; r < 8; ++r) {
                int m = m0 + mi * 16 + half * 8 + r;
                out[(size_t)m * H_ + n] = (bf16)(c[mi][ni][r] + bs);
            }
        }
}

// ---------------------------------------------------------------------------
// Kernel: att[b] = p[b] @ q_proj[b]^T   (M = LP, N = LQ, K = H), f32 out
// Same async-LDS double-buffered B staging as gemm_qproj.
// ---------------------------------------------------------------------------
__global__ void __launch_bounds__(256) gemm_att(const bf16* __restrict__ P,
                                                const bf16* __restrict__ QP,
                                                float* __restrict__ att) {
    __shared__ bf16 sB[2][64][40];
    const int b = blockIdx.z;
    const bf16* A = P + (size_t)b * LP_ * H_;
    const bf16* Bm = QP + (size_t)b * LQ_ * H_;
    float* O = att + (size_t)b * LP_ * LQ_;

    const int lane = threadIdx.x & 31;
    const int wave = threadIdx.x >> 5;
    const int lo = lane & 15, half = lane >> 4;
    const int m0 = blockIdx.x * 128 + (wave & 3) * 32;
    const int nBlock = blockIdx.y * 64;
    const int nrel0 = (wave >> 2) * 32 + lo;

    const int sn = threadIdx.x >> 2;
    const int sk = (threadIdx.x & 3) * 8;

    auto stageB = [&](int buf, int kk) {
        async_b128(Bm + (size_t)(nBlock + sn) * H_ + kk + sk, &sB[buf][sn][sk]);
    };

    stageB(0, 0);
    v8f c[2][2] = {};
    for (int kk = 0, it = 0; kk < H_; kk += 32, ++it) {
        if (kk + 32 < H_) { stageB((it + 1) & 1, kk + 32); WAIT_ASYNC(1); }
        else              { WAIT_ASYNC(0); }
        __syncthreads();
        __builtin_prefetch(A + (size_t)(m0 + lo) * H_ + kk + 64, 0, 3);
        const bf16 (&Bt)[64][40] = sB[it & 1];
        v16bf a0 = load_a_frag(A, H_, m0 + lo, kk, half);
        v16bf a1 = load_a_frag(A, H_, m0 + 16 + lo, kk, half);
        v16bf b0 = lds_b_frag_k(&Bt[nrel0][16 * half]);
        v16bf b1 = lds_b_frag_k(&Bt[nrel0 + 16][16 * half]);
        c[0][0] = WMMA_BF16(a0, b0, c[0][0]);
        c[0][1] = WMMA_BF16(a0, b1, c[0][1]);
        c[1][0] = WMMA_BF16(a1, b0, c[1][0]);
        c[1][1] = WMMA_BF16(a1, b1, c[1][1]);
        __syncthreads();
    }
    const int n0 = nBlock + (wave >> 2) * 32;
#pragma unroll
    for (int mi = 0; mi < 2; ++mi)
#pragma unroll
        for (int ni = 0; ni < 2; ++ni)
#pragma unroll
            for (int r = 0; r < 8; ++r) {
                int m = m0 + mi * 16 + half * 8 + r;
                int n = n0 + ni * 16 + lo;
                O[(size_t)m * LQ_ + n] = c[mi][ni][r];
            }
}

// ---------------------------------------------------------------------------
// Softmax statistics: one wave per row / per column (max, 1/sum(exp))
// ---------------------------------------------------------------------------
__global__ void __launch_bounds__(256) row_stats(const float* __restrict__ att,
                                                 float* __restrict__ rmax,
                                                 float* __restrict__ rrcp) {
    const int row = blockIdx.x * 8 + (threadIdx.x >> 5);
    const int lane = threadIdx.x & 31;
    const float* p = att + (size_t)row * LQ_;
    float mx = -3.0e38f;
    for (int i = lane; i < LQ_; i += 32) mx = fmaxf(mx, p[i]);
#pragma unroll
    for (int off = 16; off > 0; off >>= 1) mx = fmaxf(mx, __shfl_xor(mx, off, 32));
    float s = 0.f;
    for (int i = lane; i < LQ_; i += 32) s += __expf(p[i] - mx);
#pragma unroll
    for (int off = 16; off > 0; off >>= 1) s += __shfl_xor(s, off, 32);
    if (lane == 0) { rmax[row] = mx; rrcp[row] = 1.0f / s; }
}

__global__ void __launch_bounds__(256) col_stats(const float* __restrict__ att,
                                                 float* __restrict__ cmax,
                                                 float* __restrict__ crcp) {
    const int col = blockIdx.x * 8 + (threadIdx.x >> 5);
    const int b = col / LQ_, q = col % LQ_;
    const int lane = threadIdx.x & 31;
    const float* p = att + (size_t)b * LP_ * LQ_ + q;
    float mx = -3.0e38f;
    for (int i = lane; i < LP_; i += 32) mx = fmaxf(mx, p[(size_t)i * LQ_]);
#pragma unroll
    for (int off = 16; off > 0; off >>= 1) mx = fmaxf(mx, __shfl_xor(mx, off, 32));
    float s = 0.f;
    for (int i = lane; i < LP_; i += 32) s += __expf(p[(size_t)i * LQ_] - mx);
#pragma unroll
    for (int off = 16; off > 0; off >>= 1) s += __shfl_xor(s, off, 32);
    if (lane == 0) { cmax[col] = mx; crcp[col] = 1.0f / s; }
}

// ---------------------------------------------------------------------------
// Kernel: m_p[b] = rowsoftmax(att[b]) @ q[b]   (M = LP, N = H, K = LQ)
// B operand (q, [K=LQ, N=H]) K-major in memory -> async-stage [K][N] tile in
// LDS (double buffered), transpose on the LDS read side.
// ---------------------------------------------------------------------------
__global__ void __launch_bounds__(256) gemm_mp(const float* __restrict__ att,
                                               const bf16* __restrict__ Q,
                                               const float* __restrict__ rmax,
                                               const float* __restrict__ rrcp,
                                               float* __restrict__ mp) {
    __shared__ bf16 sB[2][32][72];
    const int b = blockIdx.z;
    const float* A = att + (size_t)b * LP_ * LQ_;
    const bf16* Qb = Q + (size_t)b * LQ_ * H_;
    float* O = mp + (size_t)b * LP_ * H_;
    const float* rmx = rmax + (size_t)b * LP_;
    const float* rrc = rrcp + (size_t)b * LP_;

    const int lane = threadIdx.x & 31;
    const int wave = threadIdx.x >> 5;
    const int lo = lane & 15, half = lane >> 4;
    const int m0 = blockIdx.x * 128 + (wave & 3) * 32;
    const int nBlock = blockIdx.y * 64;
    const int n0 = nBlock + (wave >> 2) * 32;
    const int nrel0 = (wave >> 2) * 32 + lo;

    const int row0 = m0 + lo, row1 = m0 + 16 + lo;
    const float mx0 = rmx[row0], rc0 = rrc[row0];
    const float mx1 = rmx[row1], rc1 = rrc[row1];

    const int tk = threadIdx.x >> 3;        // 0..31 (K row)
    const int tc = (threadIdx.x & 7) * 8;   // 0..56 (N chunk, 16B)

    auto stage = [&](int buf, int kk) {
        async_b128(Qb + (size_t)(kk + tk) * H_ + nBlock + tc, &sB[buf][tk][tc]);
    };

    stage(0, 0);
    v8f c[2][2] = {};
    for (int kk = 0, it = 0; kk < LQ_; kk += 32, ++it) {
        if (kk + 32 < LQ_) { stage((it + 1) & 1, kk + 32); WAIT_ASYNC(1); }
        else               { WAIT_ASYNC(0); }
        __syncthreads();
        v16bf a0 = softmax_a_frag(A + (size_t)row0 * LQ_, kk, half, mx0, rc0);
        v16bf a1 = softmax_a_frag(A + (size_t)row1 * LQ_, kk, half, mx1, rc1);
        v16bf b0 = lds_b_frag_t(sB[it & 1], nrel0, half);
        v16bf b1 = lds_b_frag_t(sB[it & 1], nrel0 + 16, half);
        c[0][0] = WMMA_BF16(a0, b0, c[0][0]);
        c[0][1] = WMMA_BF16(a0, b1, c[0][1]);
        c[1][0] = WMMA_BF16(a1, b0, c[1][0]);
        c[1][1] = WMMA_BF16(a1, b1, c[1][1]);
        __syncthreads();
    }
#pragma unroll
    for (int mi = 0; mi < 2; ++mi)
#pragma unroll
        for (int ni = 0; ni < 2; ++ni)
#pragma unroll
            for (int r = 0; r < 8; ++r) {
                int m = m0 + mi * 16 + half * 8 + r;
                int n = n0 + ni * 16 + lo;
                O[(size_t)m * H_ + n] = c[mi][ni][r];
            }
}

// ---------------------------------------------------------------------------
// Kernel: m_q[b] = colsoftmax(att[b])^T @ p[b]   (M = LQ, N = H, K = LP)
// A operand is att^T: async-stage att tile [K=32][M=128] f32 in LDS, exp on
// the transposed read. B operand (p) staged like gemm_mp. Both double buffered
// (5 async b128 per thread per stage).
// ---------------------------------------------------------------------------
__global__ void __launch_bounds__(256) gemm_mq(const float* __restrict__ att,
                                               const bf16* __restrict__ P,
                                               const float* __restrict__ cmax,
                                               const float* __restrict__ crcp,
                                               float* __restrict__ mq) {
    __shared__ float sA[2][32][132];
    __shared__ bf16 sB[2][32][72];
    const int b = blockIdx.z;
    const float* A = att + (size_t)b * LP_ * LQ_;
    const bf16* Pb = P + (size_t)b * LP_ * H_;
    float* O = mq + (size_t)b * LQ_ * H_;
    const float* cmx = cmax + (size_t)b * LQ_;
    const float* crc = crcp + (size_t)b * LQ_;

    const int lane = threadIdx.x & 31;
    const int wave = threadIdx.x >> 5;
    const int lo = lane & 15, half = lane >> 4;
    const int mBlock = blockIdx.x * 128;
    const int m0 = mBlock + (wave & 3) * 32;
    const int nBlock = blockIdx.y * 64;
    const int n0 = nBlock + (wave >> 2) * 32;
    const int nrel0 = (wave >> 2) * 32 + lo;
    const int mrel0 = (wave & 3) * 32 + lo;

    const int row0 = m0 + lo, row1 = m0 + 16 + lo;   // q indices
    const float mx0 = cmx[row0], rc0 = crc[row0];
    const float mx1 = cmx[row1], rc1 = crc[row1];

    const int tk = threadIdx.x >> 3;        // 0..31
    const int tcB = (threadIdx.x & 7) * 8;  // bf16 chunk for sB (16B)
    const int tcA = (threadIdx.x & 7) * 16; // f32 chunk for sA (4 x 16B)

    auto stage = [&](int buf, int kk) {
#pragma unroll
        for (int i = 0; i < 4; ++i)
            async_b128(A + (size_t)(kk + tk) * LQ_ + mBlock + tcA + i * 4,
                       &sA[buf][tk][tcA + i * 4]);
        async_b128(Pb + (size_t)(kk + tk) * H_ + nBlock + tcB, &sB[buf][tk][tcB]);
    };

    stage(0, 0);
    v8f c[2][2] = {};
    for (int kk = 0, it = 0; kk < LP_; kk += 32, ++it) {
        if (kk + 32 < LP_) { stage((it + 1) & 1, kk + 32); WAIT_ASYNC(5); }
        else               { WAIT_ASYNC(0); }
        __syncthreads();
        const int buf = it & 1;
        v16bf a0, a1;
#pragma unroll
        for (int e = 0; e < 16; ++e) {
            int k = 16 * half + e;
            a0[e] = (bf16)(__expf(sA[buf][k][mrel0] - mx0) * rc0);
            a1[e] = (bf16)(__expf(sA[buf][k][mrel0 + 16] - mx1) * rc1);
        }
        v16bf b0 = lds_b_frag_t(sB[buf], nrel0, half);
        v16bf b1 = lds_b_frag_t(sB[buf], nrel0 + 16, half);
        c[0][0] = WMMA_BF16(a0, b0, c[0][0]);
        c[0][1] = WMMA_BF16(a0, b1, c[0][1]);
        c[1][0] = WMMA_BF16(a1, b0, c[1][0]);
        c[1][1] = WMMA_BF16(a1, b1, c[1][1]);
        __syncthreads();
    }
#pragma unroll
    for (int mi = 0; mi < 2; ++mi)
#pragma unroll
        for (int ni = 0; ni < 2; ++ni)
#pragma unroll
            for (int r = 0; r < 8; ++r) {
                int m = m0 + mi * 16 + half * 8 + r;
                int n = n0 + ni * 16 + lo;
                O[(size_t)m * H_ + n] = c[mi][ni][r];
            }
}

// ---------------------------------------------------------------------------
// Host-side launch
// ---------------------------------------------------------------------------
extern "C" void kernel_launch(void* const* d_in, const int* in_sizes, int n_in,
                              void* d_out, int out_size, void* d_ws, size_t ws_size,
                              hipStream_t stream) {
    const float* p_enc = (const float*)d_in[0];
    const float* q_enc = (const float*)d_in[1];
    const float* g_w   = (const float*)d_in[2];
    const float* g_b   = (const float*)d_in[3];

    float* m_p = (float*)d_out;
    float* m_q = m_p + (size_t)B_ * LP_ * H_;

    char* ws = (char*)d_ws;
    const size_t np = (size_t)B_ * LP_ * H_;
    const size_t nq = (size_t)B_ * LQ_ * H_;
    const size_t ngw = (size_t)H_ * H_;
    const size_t natt = (size_t)B_ * LP_ * LQ_;

    bf16* p_bf     = (bf16*)ws;   ws += np * 2;
    bf16* q_bf     = (bf16*)ws;   ws += nq * 2;
    bf16* gw_bf    = (bf16*)ws;   ws += ngw * 2;
    bf16* qproj_bf = (bf16*)ws;   ws += nq * 2;
    float* att     = (float*)ws;  ws += natt * 4;
    float* rmax    = (float*)ws;  ws += (size_t)B_ * LP_ * 4;
    float* rrcp    = (float*)ws;  ws += (size_t)B_ * LP_ * 4;
    float* cmax    = (float*)ws;  ws += (size_t)B_ * LQ_ * 4;
    float* crcp    = (float*)ws;  ws += (size_t)B_ * LQ_ * 4;

    cast_f32_bf16<<<np / 1024, 256, 0, stream>>>(p_enc, p_bf);
    cast_f32_bf16<<<nq / 1024, 256, 0, stream>>>(q_enc, q_bf);
    cast_f32_bf16<<<ngw / 1024, 256, 0, stream>>>(g_w, gw_bf);

    gemm_qproj<<<dim3((B_ * LQ_) / 128, H_ / 64), 256, 0, stream>>>(q_bf, gw_bf, g_b, qproj_bf);

    gemm_att<<<dim3(LP_ / 128, LQ_ / 64, B_), 256, 0, stream>>>(p_bf, qproj_bf, att);

    row_stats<<<(B_ * LP_) / 8, 256, 0, stream>>>(att, rmax, rrcp);
    col_stats<<<(B_ * LQ_) / 8, 256, 0, stream>>>(att, cmax, crcp);

    gemm_mp<<<dim3(LP_ / 128, H_ / 64, B_), 256, 0, stream>>>(att, q_bf, rmax, rrcp, m_p);

    gemm_mq<<<dim3(LQ_ / 128, H_ / 64, B_), 256, 0, stream>>>(att, p_bf, cmax, crcp, m_q);

    (void)in_sizes; (void)n_in; (void)out_size; (void)ws_size;
}
